// RelationalLayer_73684458930524
// MI455X (gfx1250) — matrix operand order repaired and checked by reading, not compiled, not run on hardware
//
#include <hip/hip_runtime.h>

// ---------------------------------------------------------------------------
// RelationalLayer on gfx1250 (MI455X).
//   S  = x @ W1[0:128,:]   + b1          (41472 x 256)  bf16
//   N  = x @ W1[128:256,:]               (41472 x 256)  bf16
//   Hs[b,i,:] = sum_j relu(S[b,i,:] + N[b,edges[i,j],:])
//   out = Hs @ W2 + 20*b2                (41472 x 128)  f32
// GEMMs use V_WMMA_F32_16X16X32_BF16, bf16 operands staged in LDS.
// ---------------------------------------------------------------------------

typedef __attribute__((ext_vector_type(8)))  unsigned short v8u;
typedef __attribute__((ext_vector_type(16))) unsigned short v16u;
typedef __attribute__((ext_vector_type(16))) __bf16        v16bf;
typedef __attribute__((ext_vector_type(8)))  float          v8f;

union FragU { v16u u; v16bf b; };

__device__ __forceinline__ unsigned short f2b(float f) {
  unsigned int u = __float_as_uint(f);
  u += 0x7FFFu + ((u >> 16) & 1u);            // round-to-nearest-even
  return (unsigned short)(u >> 16);
}
__device__ __forceinline__ float b2f(unsigned short h) {
  return __uint_as_float((unsigned int)h << 16);
}

// A fragment (16x32 bf16, row-major LDS row `row`).
// Lane l: M = l&15, K-halves = kb*32 + (l>>4)*16 + {0..15}.
__device__ __forceinline__ v16bf load_a_frag(const unsigned short* row, int kb, int hi) {
  const v8u* p = (const v8u*)(row + kb * 32 + hi * 16);
  v8u lo = p[0];          // k = base .. base+7
  v8u hh = p[1];          // k = base+8 .. base+15
  FragU f;
  f.u = __builtin_shufflevector(lo, hh, 0,1,2,3,4,5,6,7,8,9,10,11,12,13,14,15);
  return f.b;
}

// B fragment (32x16 bf16). LDS holds B transposed: Bl[n][k].
// Lane l: N = l&15, K = kb*32 + (l>>4)*16 + {0..15} -> 16 contiguous halves.
__device__ __forceinline__ v16bf load_b_frag(const unsigned short* col, int kb, int hi) {
  const v8u* p = (const v8u*)(col + kb * 32 + hi * 16);
  v8u lo = p[0];
  v8u hh = p[1];
  FragU f;
  f.u = __builtin_shufflevector(lo, hh, 0,1,2,3,4,5,6,7,8,9,10,11,12,13,14,15);
  return f.b;
}

// ---------------------------------------------------------------------------
// Kernel 1: PART=0: S = X @ W1[0:128,:] + b1 -> Sp (bf16)
//           PART=1: N = X @ W1[128:256,:]    -> Nn (bf16)
// Block: 64 rows x 256 cols, K=128. 8 waves: rowtile = wave&3, colhalf = wave>>2.
// Templated so the no-bias variant carries zero bias code, and the bias load
// issues before the K-loop so it overlaps the WMMAs.
// ---------------------------------------------------------------------------
template <int PART>
__global__ __launch_bounds__(256) void k_gemm1(const float* __restrict__ X,
                                               const float* __restrict__ W1,
                                               const float* __restrict__ b1,
                                               unsigned short* __restrict__ outp)
{
  __shared__ unsigned short Al[64][136];    // 64 x 128 bf16, +8 pad (4-bank lane stride)
  __shared__ unsigned short Bl[128][136];   // transposed weight half: Bl[n][k]

  const int tid  = threadIdx.x;
  const int wave = tid >> 5, lane = tid & 31;
  const int hi   = lane >> 4, ln16 = lane & 15;
  const int rbase = blockIdx.x * 64;
  const int yoff  = PART * 128;

  // Stage A tile: 64x128 f32 -> bf16 (2048 float4, 8 per thread)
  for (int p = tid; p < 64 * 32; p += 256) {
    const int row = p >> 5;
    const int c4  = (p & 31) * 4;
    const float4 v = *(const float4*)(X + (((size_t)(rbase + row)) << 7) + c4);
    unsigned long long pk =
        (unsigned long long)f2b(v.x)
      | ((unsigned long long)f2b(v.y) << 16)
      | ((unsigned long long)f2b(v.z) << 32)
      | ((unsigned long long)f2b(v.w) << 48);
    *(unsigned long long*)&Al[row][c4] = pk;
  }
  __syncthreads();

  const int rt = wave & 3;                  // row tile
  const int cq = wave >> 2;                 // half of the staged 128 cols
  v16bf afrag[4];
  {
    const unsigned short* arow = &Al[rt * 16 + ln16][0];
    #pragma unroll
    for (int kb = 0; kb < 4; ++kb) afrag[kb] = load_a_frag(arow, kb, hi);
  }

  for (int ch = 0; ch < 2; ++ch) {          // 2 x 128-column halves of N=256
    __syncthreads();
    // Stage B half transposed: Bl[n][k] = bf16(W1[yoff+k][ch*128+n])
    for (int p = tid; p < 128 * 32; p += 256) {
      const int k  = p >> 5;
      const int n4 = (p & 31) * 4;
      const float4 w = *(const float4*)(W1 + (((size_t)(yoff + k)) << 8) + ch * 128 + n4);
      Bl[n4 + 0][k] = f2b(w.x);
      Bl[n4 + 1][k] = f2b(w.y);
      Bl[n4 + 2][k] = f2b(w.z);
      Bl[n4 + 3][k] = f2b(w.w);
    }
    __syncthreads();

    #pragma unroll
    for (int ct = 0; ct < 4; ++ct) {
      const int nt = cq * 4 + ct;           // tile within this 128-col half
      const int gn = ch * 128 + nt * 16 + ln16;
      float bias = 0.0f;
      if (PART == 0) bias = b1[gn];         // issued before the K-loop: overlaps WMMAs
      const unsigned short* bcol = &Bl[nt * 16 + ln16][0];
      v8f acc = {};
      #pragma unroll
      for (int kb = 0; kb < 4; ++kb) {
        v16bf bfrag = load_b_frag(bcol, kb, hi);
        acc = __builtin_amdgcn_wmma_f32_16x16x32_bf16(
            false, afrag[kb], false, bfrag, (short)0, acc, false, false);
      }
      #pragma unroll
      for (int r = 0; r < 8; ++r) {
        const int gm = rbase + rt * 16 + hi * 8 + r;   // C layout: M = r + 8*(lane/16)
        outp[(((size_t)gm) << 8) + gn] = f2b(acc[r] + bias);
      }
    }
  }
}

// ---------------------------------------------------------------------------
// Kernel 2: Hs[b,i,h] = sum_j relu(S[b,i,h] + N[b,edges[i,j],h]); one block/batch.
// ---------------------------------------------------------------------------
__global__ __launch_bounds__(256) void k_gather(const unsigned short* __restrict__ Sp,
                                                const unsigned short* __restrict__ Nn,
                                                const int* __restrict__ edges,
                                                unsigned short* __restrict__ Hs)
{
  __shared__ unsigned short Nl[81 * 256];   // per-batch neighbor projections (bf16)
  __shared__ int eds[81 * 20];

  const int b = blockIdx.x, tid = threadIdx.x;
  const unsigned short* Nb = Nn + (size_t)b * 81 * 256;

  for (int p = tid; p < (81 * 256) / 8; p += 256)
    *(v8u*)&Nl[p * 8] = *(const v8u*)(Nb + p * 8);
  for (int p = tid; p < 81 * 20; p += 256) eds[p] = edges[p];
  __syncthreads();

  const int h = tid;                         // H == 256 == blockDim
  for (int i = 0; i < 81; ++i) {
    const float s = b2f(Sp[((size_t)b * 81 + i) * 256 + h]);   // already has +b1
    const int* e = &eds[i * 20];
    float acc = 0.0f;
    #pragma unroll
    for (int j = 0; j < 20; ++j) {
      const float v = s + b2f(Nl[e[j] * 256 + h]);
      acc += fmaxf(v, 0.0f);
    }
    Hs[((size_t)b * 81 + i) * 256 + h] = f2b(acc);
  }
}

// ---------------------------------------------------------------------------
// Kernel 3: out = Hs(41472x256 bf16) @ W2(256x128) + 20*b2, f32 out.
// Block: 32 rows. 8 waves: rowtile = wave&1, coltile(within 64-col half) = wave>>1.
// ---------------------------------------------------------------------------
__global__ __launch_bounds__(256) void k_gemm3(const unsigned short* __restrict__ Hs,
                                               const float* __restrict__ W2,
                                               const float* __restrict__ b2,
                                               float* __restrict__ Out)
{
  __shared__ unsigned short Al[32][264];    // 32 x 256 bf16 + pad
  __shared__ unsigned short Bl[64][264];    // transposed W2 half: Bl[n][k]

  const int tid  = threadIdx.x;
  const int wave = tid >> 5, lane = tid & 31;
  const int hi   = lane >> 4, ln16 = lane & 15;
  const int rbase = blockIdx.x * 32;

  // Stage A: already bf16, 16B chunks
  for (int p = tid; p < 32 * 32; p += 256) {
    const int row = p >> 5;
    const int kc  = (p & 31) * 8;
    *(v8u*)&Al[row][kc] = *(const v8u*)(Hs + (((size_t)(rbase + row)) << 8) + kc);
  }
  __syncthreads();

  const int rt = wave & 1;
  const int nt = wave >> 1;                 // 0..3 within 64-col half
  v16bf afrag[8];
  {
    const unsigned short* arow = &Al[rt * 16 + ln16][0];
    #pragma unroll
    for (int kb = 0; kb < 8; ++kb) afrag[kb] = load_a_frag(arow, kb, hi);
  }

  for (int ch = 0; ch < 2; ++ch) {          // 2 x 64-column halves of N=128
    __syncthreads();
    for (int p = tid; p < 256 * 16; p += 256) {
      const int k  = p >> 4;
      const int n4 = (p & 15) * 4;
      const float4 w = *(const float4*)(W2 + (((size_t)k) << 7) + ch * 64 + n4);
      Bl[n4 + 0][k] = f2b(w.x);
      Bl[n4 + 1][k] = f2b(w.y);
      Bl[n4 + 2][k] = f2b(w.z);
      Bl[n4 + 3][k] = f2b(w.w);
    }
    __syncthreads();

    const int gn = ch * 64 + nt * 16 + ln16;
    const float bias = 20.0f * b2[gn];      // neighbor-sum folds 20 copies of b2
    const unsigned short* bcol = &Bl[nt * 16 + ln16][0];
    v8f acc = {};
    #pragma unroll
    for (int kb = 0; kb < 8; ++kb) {
      v16bf bfrag = load_b_frag(bcol, kb, hi);
      acc = __builtin_amdgcn_wmma_f32_16x16x32_bf16(
          false, afrag[kb], false, bfrag, (short)0, acc, false, false);
    }
    #pragma unroll
    for (int r = 0; r < 8; ++r) {
      const int gm = rbase + rt * 16 + hi * 8 + r;
      Out[(((size_t)gm) << 7) + gn] = acc[r] + bias;
    }
  }
}

// ---------------------------------------------------------------------------
extern "C" void kernel_launch(void* const* d_in, const int* in_sizes, int n_in,
                              void* d_out, int out_size, void* d_ws, size_t ws_size,
                              hipStream_t stream) {
  const float* x     = (const float*)d_in[0];   // (512,81,128)
  const int*   edges = (const int*)  d_in[1];   // (81,20)
  const float* W1    = (const float*)d_in[2];   // (256,256)
  const float* b1    = (const float*)d_in[3];   // (256)
  const float* W2    = (const float*)d_in[4];   // (256,128)
  const float* b2    = (const float*)d_in[5];   // (128)
  float* out = (float*)d_out;                   // (512,81,128) f32

  const size_t RH = (size_t)41472 * 256;        // halves per intermediate buffer
  unsigned short* Sp = (unsigned short*)d_ws;   // S + b1          (bf16)
  unsigned short* Nn = Sp + RH;                 // neighbor proj   (bf16)
  unsigned short* Hs = Nn + RH;                 // relu-summed h   (bf16)

  k_gemm1<0><<<41472 / 64, 256, 0, stream>>>(x, W1, b1, Sp);
  k_gemm1<1><<<41472 / 64, 256, 0, stream>>>(x, W1, b1, Nn);
  k_gather <<<512,         256, 0, stream>>>(Sp, Nn, edges, Hs);
  k_gemm3  <<<41472 / 32,  256, 0, stream>>>(Hs, W2, b2, out);
}
